// Model_29652454212339
// MI455X (gfx1250) — matrix operand order repaired and checked by reading, not compiled, not run on hardware
//
#include <hip/hip_runtime.h>
#include <stdint.h>

#define BB 64
#define NN 2048
#define NODES (BB * NN)

typedef __attribute__((ext_vector_type(16))) _Float16 v16h;
typedef __attribute__((ext_vector_type(8)))  _Float16 half8;
typedef __attribute__((ext_vector_type(8)))  float    v8f;
typedef __attribute__((ext_vector_type(4)))  unsigned int v4u;
typedef __attribute__((ext_vector_type(8)))  int      v8i;
typedef __attribute__((ext_vector_type(4)))  int      v4i;

#if defined(__has_builtin)
#  if __has_builtin(__builtin_amdgcn_tensor_load_to_lds)
#    define HAVE_TDM 1
#  else
#    define HAVE_TDM 0
#  endif
#else
#  define HAVE_TDM 0
#endif

__device__ __forceinline__ half8 ld8g(const _Float16* p) {
    return *(const half8*)p;
}

__device__ __forceinline__ v16h make_a(half8 lo, half8 hi) {
    v16h a;
#pragma unroll
    for (int e = 0; e < 8; ++e) { a[e] = lo[e]; a[e + 8] = hi[e]; }
    return a;
}

// A fragment (16x32 f16): lane L -> row (L&15); khalf=L>>4 selects K runs
__device__ __forceinline__ v16h a_frag_global(const _Float16* row, int kbase, int khalf) {
    return make_a(ld8g(row + kbase + 8 * khalf),
                  ld8g(row + kbase + 16 + 8 * khalf));
}

__device__ __forceinline__ v8f wmma_f16(v16h a, v16h b, v8f c) {
    return __builtin_amdgcn_wmma_f32_16x16x32_f16(false, a, false, b, (short)0, c, false, false);
}

// packed B fragment (from LDS staging): frag f, lane -> 16 contiguous halfs
__device__ __forceinline__ v16h b_frag_lds(const _Float16* wp, int f, int lane) {
    return *(const v16h*)(wp + ((size_t)(f * 32 + lane) << 4));
}

// ---- TDM: 1-D tile (ndw dwords) global -> LDS. D# per CDNA5 ISA ch.8 bit layout.
__device__ __forceinline__ void tdm_copy_1d(const void* gsrc, unsigned lds_off, unsigned ndw) {
#if HAVE_TDM
    const unsigned long long ga = (unsigned long long)(uintptr_t)gsrc;
    v4u g0;
    g0[0] = 1u;                                            // count=1 (valid user D#)
    g0[1] = lds_off;                                       // lds_addr (bytes)
    g0[2] = (unsigned)(ga & 0xffffffffu);                  // global_addr[31:0]
    g0[3] = (unsigned)((ga >> 32) & 0x1ffffffu) | (2u << 30); // global_addr[56:32] | type=2
    v8i g1;
    g1[0] = (int)(2u << 16);                               // data_size=2 (4B), wg_mask=0
    g1[1] = (int)((ndw & 0xffffu) << 16);                  // tensor_dim0[15:0]
    g1[2] = (int)(((ndw >> 16) & 0xffffu) | (1u << 16));   // tensor_dim0[31:16] | tensor_dim1=1
    g1[3] = (int)((ndw & 0xffffu) << 16);                  // tile_dim0 = ndw
    g1[4] = 1;                                             // tile_dim1=1, tile_dim2=0
    g1[5] = (int)ndw;                                      // tensor_dim0_stride[31:0]
    g1[6] = 0;
    g1[7] = 0;
    v4i z4 = {0, 0, 0, 0};
    v8i z8 = {0, 0, 0, 0, 0, 0, 0, 0};
    __builtin_amdgcn_tensor_load_to_lds(g0, g1, z4, z4, z8, 0);
    __builtin_amdgcn_s_wait_tensorcnt(0);
#else
    (void)gsrc; (void)lds_off; (void)ndw;
#endif
}

// stage `ndw` dwords of packed weights into LDS (wave0 issues TDM, all waves barrier)
__device__ __forceinline__ void stage_weights(const _Float16* __restrict__ gsrc,
                                              _Float16* lds_dst, unsigned ndw,
                                              int wave, int lane) {
#if HAVE_TDM
    if (wave == 0) {
        tdm_copy_1d(gsrc, (unsigned)(uintptr_t)lds_dst, ndw);
    }
#else
    // cooperative copy: 256 threads, b128 chunks
    const int t = wave * 32 + lane;
    const int n128 = (int)(ndw / 4);
    for (int i = t; i < n128; i += 256)
        ((half8*)lds_dst)[i] = ((const half8*)gsrc)[i];
#endif
    __syncthreads();
}

// ---------------- one-shot weight packer: B layout = lane(L&15)=col, khalf=L>>4, elem e -> K=32c+8*khalf+(e&7)+16*(e>>3)
__global__ void pack_w(const float* __restrict__ W, int Kreal, int Ncols,
                       int nkc, int nct, _Float16* __restrict__ dst) {
    const int tid = blockIdx.x * blockDim.x + threadIdx.x;
    const int total = nkc * nct * 32;
    if (tid >= total) return;
    const int lane = tid & 31;
    const int f = tid >> 5;
    const int c = f / nct;
    const int ct = f - c * nct;
    const int khalf = lane >> 4;
    const int col = ct * 16 + (lane & 15);
    _Float16* out = dst + ((size_t)tid << 4);
#pragma unroll
    for (int e = 0; e < 16; ++e) {
        const int K = 32 * c + 8 * khalf + (e & 7) + 16 * (e >> 3);
        float v = 0.0f;
        if (K < Kreal && col < Ncols) v = W[K * Ncols + col];
        out[e] = (_Float16)v;
    }
}

__global__ void pack_bias(const float* __restrict__ bf, int FOUT, int npad,
                          float* __restrict__ dst) {
    const int tid = blockIdx.x * blockDim.x + threadIdx.x;
    if (tid >= npad) return;
    dst[tid] = (tid < FOUT) ? bf[tid] : 0.0f;
}

// ---------------- prep: h = concat[frames7, features, posenc, t/100], f16, K padded to 64
__global__ void prep_kernel(const float* __restrict__ frames7,
                            const float* __restrict__ features,
                            const int* __restrict__ tptr,
                            _Float16* __restrict__ hbuf) {
    int idx = blockIdx.x * blockDim.x + threadIdx.x;
    if (idx >= NODES) return;
    int n = idx & (NN - 1);
    _Float16* row = hbuf + (size_t)idx * 64;
    const float* f7 = frames7 + (size_t)idx * 7;
    const float* fe = features + (size_t)idx * 16;
#pragma unroll
    for (int k = 0; k < 7; ++k) row[k] = (_Float16)f7[k];
#pragma unroll
    for (int k = 0; k < 16; ++k) row[7 + k] = (_Float16)fe[k];
    const float fn = (float)n;
#pragma unroll
    for (int d = 0; d < 16; ++d) {
        const int i2 = d & ~1;
        const float div = __expf(-0.5756462732485115f * (float)i2); // -ln(10000)/16
        const float ang = fn * div;
        row[23 + d] = (_Float16)((d & 1) ? __cosf(ang) : __sinf(ang));
    }
    row[39] = (_Float16)((float)(*tptr) * 0.01f);
#pragma unroll
    for (int k = 40; k < 64; ++k) row[k] = (_Float16)0.0f;
}

// ---------------- msg: miT = (h@Wl)^T, mjT = (h@Wr)^T, stored transposed [32][NODES]
// wlr = packed Wl(4 frags) | Wr(4 frags) contiguous (8 KB), TDM-staged to LDS.
template <int HIN>
__global__ void __launch_bounds__(256) msg_kernel(const _Float16* __restrict__ hbuf,
                                                  const _Float16* __restrict__ wlr,
                                                  _Float16* __restrict__ miT,
                                                  _Float16* __restrict__ mjT) {
    const int lane = threadIdx.x & 31;
    const int wave = threadIdx.x >> 5;
    const int tile = blockIdx.x * 8 + wave;
    const int tpb = NN / 16;
    const int b = tile / tpb;
    const int n0 = (tile - b * tpb) * 16;
    const size_t base = (size_t)b * NN + n0;
    const int khalf = lane >> 4;
    const int colb = lane & 15;

    __shared__ __align__(32) _Float16 smw[8 * 512];   // 8 frags (Wl:0..3, Wr:4..7)
    stage_weights(wlr, smw, (8 * 512 * 2) / 4, wave, lane);

    const _Float16* row = hbuf + (base + (lane & 15)) * 64;
    const v16h a0 = a_frag_global(row, 0, khalf);
    const v16h a1 = a_frag_global(row, 32, khalf);

#pragma unroll
    for (int m = 0; m < 2; ++m) {
        _Float16* dst = m ? mjT : miT;
#pragma unroll
        for (int ct = 0; ct < 2; ++ct) {
            const int col = ct * 16 + colb;
            v8f acc = {};
            acc = wmma_f16(a0, b_frag_lds(smw, m * 4 + 0 * 2 + ct, lane), acc);
            acc = wmma_f16(a1, b_frag_lds(smw, m * 4 + 1 * 2 + ct, lane), acc);
            // transposed store: lane's 8 elements are 8 consecutive nodes at fixed col
            half8 st;
#pragma unroll
            for (int r = 0; r < 8; ++r) st[r] = (_Float16)acc[r];
            *(half8*)(dst + (size_t)col * NODES + base + 8 * khalf) = st;
        }
    }
}

// ---------------- feat: msum (masked +-1 neighbor sum) then concat[h,msum] @ Wf + bf
template <int HIN, int FOUT, bool RELU, bool QUAT>
__global__ void __launch_bounds__(256) feat_kernel(
    const _Float16* __restrict__ hbuf,       // stride 64
    const _Float16* __restrict__ miT,        // [32][NODES]
    const _Float16* __restrict__ mjT,        // [32][NODES]
    const unsigned char* __restrict__ mask,  // node_mask (bool)
    const float* __restrict__ bmsg,          // 32
    const _Float16* __restrict__ wfp,        // packed Wf fragments (3*NCT frags)
    const float* __restrict__ bfp,           // bias padded to NCT*16
    _Float16* __restrict__ outh,             // !QUAT: stride 64
    float* __restrict__ outf)                // QUAT: stride 7
{
    constexpr int KCAT = HIN + 32;           // 72 or 96 (real concat width)
    constexpr int NCT = (FOUT + 15) / 16;
    constexpr int NFRAG = 3 * NCT;
    const int lane = threadIdx.x & 31;
    const int wave = threadIdx.x >> 5;
    const int tile = blockIdx.x * 8 + wave;
    const int tpb = NN / 16;
    const int b = tile / tpb;
    const int n0 = (tile - b * tpb) * 16;
    const size_t base = (size_t)b * NN + n0;

    __shared__ __align__(32) _Float16 smw[NFRAG * 512];
    __shared__ __align__(16) _Float16 msumLDS[8][16][32];

    stage_weights(wfp, smw, (NFRAG * 512 * 2) / 4, wave, lane);

    // masked neighbor message sum: lane = column (transposed mi/mj => contiguous node runs)
    {
        const int c = lane;
        const float bm = bmsg[c];
        const _Float16* mic = miT + (size_t)c * NODES + base;
        const _Float16* mjc = mjT + (size_t)c * NODES + base;
        const unsigned char* mk = mask + base;
#pragma unroll
        for (int r = 0; r < 16; ++r) {
            const int n = n0 + r;
            const float miv = (float)mic[r];
            const float nmv = (float)mk[r];
            float prev = 0.f, pm = 0.f, nxt = 0.f, nmk = 0.f;
            if (n > 0)      { prev = (float)mjc[r - 1]; pm  = (float)mk[r - 1]; }
            if (n < NN - 1) { nxt  = (float)mjc[r + 1]; nmk = (float)mk[r + 1]; }
            const float ms = (miv + prev + bm) * (nmv * pm) + (miv + nxt + bm) * (nmv * nmk);
            msumLDS[wave][r][c] = (_Float16)ms;
        }
    }
    __syncthreads();

    const int khalf = lane >> 4;
    const int colb = lane & 15;
    const int arow = lane & 15;
    const _Float16* hrow = hbuf + (base + arow) * 64;

    v8f acc[NCT];
#pragma unroll
    for (int ct = 0; ct < NCT; ++ct) {
        const float bv = bfp[ct * 16 + colb];
#pragma unroll
        for (int r = 0; r < 8; ++r) acc[ct][r] = bv;
    }

#pragma unroll
    for (int c = 0; c < 3; ++c) {
        half8 run[2];
#pragma unroll
        for (int u = 0; u < 2; ++u) {
            // 8-aligned K runs: each run comes entirely from h, msum, or zero pad
            const int kb = 32 * c + 16 * u + 8 * khalf;
            if (kb < HIN) {
                run[u] = ld8g(hrow + kb);
            } else if (kb < KCAT) {
                run[u] = *(const half8*)&msumLDS[wave][arow][kb - HIN];
            } else {
                half8 z = {};
                run[u] = z;
            }
        }
        const v16h a = make_a(run[0], run[1]);
#pragma unroll
        for (int ct = 0; ct < NCT; ++ct)
            acc[ct] = wmma_f16(a, b_frag_lds(smw, c * NCT + ct, lane), acc[ct]);
    }

    if (!QUAT) {
#pragma unroll
        for (int ct = 0; ct < NCT; ++ct) {
            const int col = ct * 16 + colb;
#pragma unroll
            for (int r = 0; r < 8; ++r) {
                float v = acc[ct][r];
                if (RELU) v = v > 0.f ? v : 0.f;
                outh[(base + r + 8 * khalf) * 64 + col] = (_Float16)v;
            }
        }
    } else {
        // quaternion normalize: cols 0..3 live in lanes {0..3} / {16..19}; xor-reduce within group of 4
#pragma unroll
        for (int r = 0; r < 8; ++r) {
            const float v = acc[0][r];
            float s = (colb < 4) ? v * v : 0.f;
            s += __shfl_xor(s, 1, 32);
            s += __shfl_xor(s, 2, 32);
            const float outv = (colb < 4) ? v / sqrtf(s) : v;
            if (colb < 7)
                outf[(base + r + 8 * khalf) * 7 + colb] = outv;
        }
    }
}

extern "C" void kernel_launch(void* const* d_in, const int* in_sizes, int n_in,
                              void* d_out, int out_size, void* d_ws, size_t ws_size,
                              hipStream_t stream) {
    const float* frames7  = (const float*)d_in[0];
    const float* features = (const float*)d_in[1];
    const unsigned char* mask = (const unsigned char*)d_in[2];
    const float* Wm1 = (const float*)d_in[3];
    const float* bm1 = (const float*)d_in[4];
    const float* Wf1 = (const float*)d_in[5];
    const float* bf1 = (const float*)d_in[6];
    const float* Wm2 = (const float*)d_in[7];
    const float* bm2 = (const float*)d_in[8];
    const float* Wf2 = (const float*)d_in[9];
    const float* bf2 = (const float*)d_in[10];
    const int* tptr  = (const int*)d_in[11];
    float* out = (float*)d_out;

    // workspace layout (f16): h | i | miT | mjT | packed weights/bias
    _Float16* hbuf = (_Float16*)d_ws;
    _Float16* ibuf = hbuf + (size_t)NODES * 64;
    _Float16* miT  = ibuf + (size_t)NODES * 64;
    _Float16* mjT  = miT  + (size_t)NODES * 32;
    _Float16* w1p  = mjT  + (size_t)NODES * 32;   // Wl1|Wr1: 8 frags * 512 halfs
    _Float16* wf1p = w1p  + 8 * 512;              // 12 frags
    _Float16* w2p  = wf1p + 12 * 512;             // Wl2|Wr2: 8 frags
    _Float16* wf2p = w2p  + 8 * 512;              // 3 frags
    float*    bf1p = (float*)(wf2p + 3 * 512);    // 64 floats
    float*    bf2p = bf1p + 64;                   // 16 floats

    const int tiles = NODES / 16;     // 8192 wave-tiles
    const int blocks = tiles / 8;     // 8 waves (256 threads) per block

    prep_kernel<<<(NODES + 255) / 256, 256, 0, stream>>>(frames7, features, tptr, hbuf);

    // one-shot weight/bias packing (lane-ordered f16 fragments)
    pack_w<<<2, 64, 0, stream>>>(Wm1,           40, 32, 2, 2, w1p);
    pack_w<<<2, 64, 0, stream>>>(Wm1 + 40 * 32, 40, 32, 2, 2, w1p + 4 * 512);
    pack_w<<<6, 64, 0, stream>>>(Wf1,           72, 64, 3, 4, wf1p);
    pack_w<<<2, 64, 0, stream>>>(Wm2,           64, 32, 2, 2, w2p);
    pack_w<<<2, 64, 0, stream>>>(Wm2 + 64 * 32, 64, 32, 2, 2, w2p + 4 * 512);
    pack_w<<<2, 64, 0, stream>>>(Wf2,           96,  7, 3, 1, wf2p);
    pack_bias<<<1, 64, 0, stream>>>(bf1, 64, 64, bf1p);
    pack_bias<<<1, 16, 0, stream>>>(bf2,  7, 16, bf2p);

    msg_kernel<40><<<blocks, 256, 0, stream>>>(hbuf, w1p, miT, mjT);
    feat_kernel<40, 64, true, false><<<blocks, 256, 0, stream>>>(hbuf, miT, mjT, mask, bm1, wf1p, bf1p, ibuf, nullptr);
    msg_kernel<64><<<blocks, 256, 0, stream>>>(ibuf, w2p, miT, mjT);
    feat_kernel<64, 7, false, true><<<blocks, 256, 0, stream>>>(ibuf, miT, mjT, mask, bm2, wf2p, bf2p, nullptr, out);
}
